// RotorLayer_38465727103157
// MI455X (gfx1250) — matrix operand order repaired and checked by reading, not compiled, not run on hardware
//
#include <hip/hip_runtime.h>
#include <hip/hip_bf16.h>

// out = tanh(X @ W^T + b):  M=131072, N=K=256, fp32 in/out.
// Memory-bound (268 MB @ 23.3 TB/s ~= 11.5us floor; AI ~ 64 F/B) -> f16 WMMA
// (v_wmma_f32_16x16x32_f16, fp32 accumulate) keeps compute above the
// ~1.5 PFLOPS needed to sit on the HBM roof. W lives in LDS as f16 (staged
// once per WG; fp32 copy stays L2-resident), B fragments persist in VGPRs.
// X staging is software-pipelined through a double-buffered LDS tile; the
// epilogue uses gfx1250 v_tanh_f32.

typedef __attribute__((ext_vector_type(16))) _Float16 v16h;
typedef __attribute__((ext_vector_type(8)))  float    v8f;

#define HID 256
#define SXS 264                 // padded LDS row stride (halfs): 528B -> 4-bank row shift
#define MTILES_PER_WG 16
#define ROWS_PER_WG (MTILES_PER_WG * 16)

union V16H { v16h v; uint4 q[2]; };

__device__ __forceinline__ float fast_tanh(float v) {
#if __has_builtin(__builtin_amdgcn_tanhf)
    return __builtin_amdgcn_tanhf(v);          // v_tanh_f32 (gfx1250 TRANS op)
#elif __has_builtin(__builtin_amdgcn_tanh_f32)
    return __builtin_amdgcn_tanh_f32(v);
#else
    return tanhf(v);
#endif
}

__global__ __launch_bounds__(256)
void rotor_tanh_gemm(const float* __restrict__ x,
                     const float* __restrict__ W,
                     const float* __restrict__ bias,
                     float* __restrict__ out, int M)
{
    extern __shared__ _Float16 smem[];
    _Float16* sW = smem;                        // [256][SXS] W rows as f16
    _Float16* sX0 = smem + HID * SXS;           // [16][SXS]  X tile buf 0
    _Float16* sX1 = sX0 + 16 * SXS;             // [16][SXS]  X tile buf 1

    const int tid  = threadIdx.x;
    const int wave = tid >> 5;                  // 8 waves per block (wave32)
    const int lane = tid & 31;
    const int hv   = lane >> 4;                 // lane half (0: 0-15, 1: 16-31)
    const int r    = lane & 15;

    // ---- Stage W (fp32 -> f16) into LDS once per WG.
    for (int i = tid; i < HID * (HID / 4); i += 256) {
        const int row = i >> 6;
        const int c4  = i & 63;
        float4 w = reinterpret_cast<const float4*>(W)[row * (HID / 4) + c4];
        _Float16* dst = sW + row * SXS + c4 * 4;
        dst[0] = (_Float16)w.x; dst[1] = (_Float16)w.y;
        dst[2] = (_Float16)w.z; dst[3] = (_Float16)w.w;
    }

    // Per-thread staging geometry: i = tid + 256*j -> row = (tid>>6)+4j, col4 fixed.
    const int st_row = tid >> 6;                // base row (0..3), +4 per j
    const int st_c4  = tid & 63;                // float4 column (0..63)
    const int m_wg   = blockIdx.x * ROWS_PER_WG;

    // ---- Stage X tile 0 directly into buf 0.
#pragma unroll
    for (int j = 0; j < 4; ++j) {
        const int row = st_row + 4 * j;
        const int m   = m_wg + row;
        const float4* src = reinterpret_cast<const float4*>(x)
                            + (size_t)m * (HID / 4) + st_c4;
        float4 xv = (m < M) ? *src : make_float4(0.f, 0.f, 0.f, 0.f);
        _Float16* dst = sX0 + row * SXS + st_c4 * 4;
        dst[0] = (_Float16)xv.x; dst[1] = (_Float16)xv.y;
        dst[2] = (_Float16)xv.z; dst[3] = (_Float16)xv.w;
    }
    __syncthreads();

    // ---- This wave owns output columns [n_base, n_base+32).
    const int n_base = wave * 32;
    const float b0 = bias[n_base + r];
    const float b1 = bias[n_base + 16 + r];

    // ---- Persistent B fragments (contiguous k order, same as A: lane half h holds
    // K=16h..16h+15 per 32-K chunk, 2 K per VGPR -> 16 contiguous halfs of W row n per lane).
    v16h Bf[16];
#pragma unroll
    for (int nt = 0; nt < 2; ++nt) {
#pragma unroll
        for (int kc = 0; kc < 8; ++kc) {
            const _Float16* p = sW + (n_base + nt * 16 + r) * SXS + kc * 32 + hv * 16;
            V16H t;
            t.q[0] = *reinterpret_cast<const uint4*>(p);
            t.q[1] = *reinterpret_cast<const uint4*>(p + 8);
            Bf[nt * 8 + kc] = t.v;
        }
    }

    for (int mt = 0; mt < MTILES_PER_WG; ++mt) {
        const int m0 = m_wg + mt * 16;
        const _Float16* sXc = (mt & 1) ? sX1 : sX0;   // compute buffer
        _Float16*       sXn = (mt & 1) ? sX0 : sX1;   // next-tile buffer
        const bool have_next = (mt + 1 < MTILES_PER_WG);

        // ---- Issue next tile's global loads early (latency overlaps WMMA).
        float4 pre[4];
        if (have_next) {
            const int m0n = m0 + 16;
#pragma unroll
            for (int j = 0; j < 4; ++j) {
                const int m = m0n + st_row + 4 * j;
                const float4* src = reinterpret_cast<const float4*>(x)
                                    + (size_t)m * (HID / 4) + st_c4;
                pre[j] = (m < M) ? *src : make_float4(0.f, 0.f, 0.f, 0.f);
                // gfx1250 speculative prefetch of the tile after next
                __builtin_prefetch(reinterpret_cast<const float*>(src) + 16 * HID, 0, 1);
            }
        }

        // ---- Preload ALL A fragments first (16 ds_load_b128 as one clause)
        // so the WMMA chain only needs partial s_wait_dscnt waits.
        // 16-bit A 16x32 layout (matches B): lane half h holds K = 16h..16h+15.
        const _Float16* pa_base = sXc + r * SXS + hv * 16;
        V16H a[8];
#pragma unroll
        for (int kc = 0; kc < 8; ++kc) {
            const _Float16* pa = pa_base + kc * 32;
            a[kc].q[0] = *reinterpret_cast<const uint4*>(pa);
            a[kc].q[1] = *reinterpret_cast<const uint4*>(pa + 8);
        }

        v8f acc0 = {};
        v8f acc1 = {};
#pragma unroll
        for (int kc = 0; kc < 8; ++kc) {
            acc0 = __builtin_amdgcn_wmma_f32_16x16x32_f16(
                false, a[kc].v, false, Bf[kc],     (short)0, acc0, false, false);
            acc1 = __builtin_amdgcn_wmma_f32_16x16x32_f16(
                false, a[kc].v, false, Bf[8 + kc], (short)0, acc1, false, false);
        }

        // ---- Epilogue: C/D layout VGPR i -> M = i + 8*hv, N = lane%16.
        // One 64-bit base per tile; stores at constant immediate offsets.
        float* po = out + (size_t)(m0 + 8 * hv) * HID + n_base + r;
        if (m0 + 16 <= M) {
#pragma unroll
            for (int i = 0; i < 8; ++i) {
                po[i * HID]      = fast_tanh(acc0[i] + b0);
                po[i * HID + 16] = fast_tanh(acc1[i] + b1);
            }
        } else {
#pragma unroll
            for (int i = 0; i < 8; ++i) {
                if (m0 + i + 8 * hv < M) {
                    po[i * HID]      = fast_tanh(acc0[i] + b0);
                    po[i * HID + 16] = fast_tanh(acc1[i] + b1);
                }
            }
        }

        // ---- Convert + store the prefetched tile into the other buffer.
        if (have_next) {
#pragma unroll
            for (int j = 0; j < 4; ++j) {
                const int row = st_row + 4 * j;
                _Float16* dst = sXn + row * SXS + st_c4 * 4;
                dst[0] = (_Float16)pre[j].x; dst[1] = (_Float16)pre[j].y;
                dst[2] = (_Float16)pre[j].z; dst[3] = (_Float16)pre[j].w;
            }
        }
        __syncthreads();
    }
}

extern "C" void kernel_launch(void* const* d_in, const int* in_sizes, int n_in,
                              void* d_out, int out_size, void* d_ws, size_t ws_size,
                              hipStream_t stream) {
    const float* x    = (const float*)d_in[0];   // (16, 8192, 256) fp32
    const float* W    = (const float*)d_in[1];   // (256, 256) fp32
    const float* bias = (const float*)d_in[2];   // (256,) fp32
    float* out = (float*)d_out;

    const int M = in_sizes[0] / HID;             // 131072 flattened rows
    const int nblocks = (M + ROWS_PER_WG - 1) / ROWS_PER_WG;   // 512
    const size_t lds_bytes = (size_t)(HID + 32) * SXS * sizeof(_Float16); // ~152 KB

    rotor_tanh_gemm<<<dim3(nblocks), dim3(256), lds_bytes, stream>>>(x, W, bias, out, M);
}
